// CombinedSegmentationLoss_2413771620741
// MI455X (gfx1250) — compile-verified, matrix operand order checked by reading, not run on hardware
//
#include <hip/hip_runtime.h>
#include <hip/hip_bf16.h>

typedef __attribute__((ext_vector_type(16))) _Float16 v16h;
typedef __attribute__((ext_vector_type(8)))  float    v8f;
typedef __attribute__((ext_vector_type(2)))  __fp16   fp16x2;

#define NC       88              // classes
#define SPATIAL  (96*96*96)      // 884736 voxels per batch item
#define CBYTES   ((size_t)SPATIAL * 4)   // bytes per channel plane (3538944)
#define NB       2               // batch
#define BLK      128             // 4 waves (wave32)
#define GX       864             // x-blocks per batch item
#define VSWEEP   (2*GX*BLK)      // 221184 voxels per grid sweep
#define ITERS    4               // SPATIAL / VSWEEP, exact
#define RSTRIDE  68              // padded LDS row stride (dwords) per channel-pair
#define SMOOTH   1e-5f
#define DICE_W   0.6f
#define CE_W     0.4f

// workspace layout (floats): [0,176) pred_o  [176,352) inter  [352,528) ground
//                            [528] nll_sum   [529] valid_count
#define WS_PREDO  0
#define WS_INTER  176
#define WS_GROUND 352
#define WS_NLL    528
#define WS_VCNT   529

static __device__ __forceinline__ unsigned pk16(float a, float b) {
  union { fp16x2 h; unsigned u; } r;
  r.h = __builtin_amdgcn_cvt_pkrtz(a, b);   // v_cvt_pk_rtz_f16_f32
  return r.u;
}

// Pin a uniform pointer into an SGPR pair and make it opaque, so loads select
// the GLOBAL_LOAD*_SADDR form: scalar base + one shared 32-bit VGPR offset.
static __device__ __forceinline__ const char* sbase(const char* p) {
  asm("" : "+s"(p));
  return p;
}

__global__ __launch_bounds__(BLK)
void loss_main(const float* __restrict__ pred, const int* __restrict__ target,
               float* __restrict__ ws) {
  // per-wave f16 e^2 tile: 48 channel-pair rows x 64 voxels (+4 pad dwords/row)
  __shared__ __align__(16) unsigned int tile[4][48 * RSTRIDE];  // ~51 KB
  // per-wave inv(s)^2 per voxel, f16 pair-packed: dword j = voxels (2j, 2j+1)
  __shared__ __align__(16) unsigned int sinvd[4][32];
  __shared__ float s_inter[NC];
  __shared__ float s_ground[NC];
  __shared__ float s_predo[NC];
  __shared__ float s_nll;
  __shared__ float s_vcnt;

  const int tid  = threadIdx.x;
  const int wave = tid >> 5;
  const int lane = tid & 31;
  const int col  = lane & 15;        // WMMA column N = lane % 16
  const int hsel = lane >> 4;        // which 16-voxel run this lane's fragment holds
  const int b    = blockIdx.y;

  if (tid < NC) { s_inter[tid] = 0.f; s_ground[tid] = 0.f; s_predo[tid] = 0.f; }
  if (tid == 0) { s_nll = 0.f; s_vcnt = 0.f; }
  unsigned int* wt = &tile[wave][0];
  // zero padding channel-pair rows 44..47 (channels 88..95) once
  #pragma unroll
  for (int p = 44; p < 48; ++p) { wt[p * RSTRIDE + lane] = 0u; wt[p * RSTRIDE + 32 + lane] = 0u; }
  __syncthreads();

  v8f cf[6];
  #pragma unroll
  for (int t = 0; t < 6; ++t) cf[t] = (v8f){0.f,0.f,0.f,0.f,0.f,0.f,0.f,0.f};
  v16h ones;
  #pragma unroll
  for (int i = 0; i < 16; ++i) ones[i] = (_Float16)1.0f;

  float nll_loc = 0.f, vcnt_loc = 0.f;
  // uniform bases (SGPRs); the divergent part is a single 32-bit BYTE offset
  const char* __restrict__ pB = (const char*)(pred + (size_t)b * NC * SPATIAL);
  const char* __restrict__ tB = (const char*)(target + (size_t)b * SPATIAL);

  for (int it = 0; it < ITERS; ++it) {
    // this thread owns adjacent voxels 2*idx0, 2*idx0+1
    const unsigned idx0 = (unsigned)(blockIdx.x * BLK + tid) + (unsigned)it * (VSWEEP / 2);
    const unsigned boff = idx0 * 8u;                     // byte offset (u32 -> SADDR voffset)
    const int2 tg = *(const int2*)(sbase(tB) + boff);
    const bool va_ok = (tg.x != -1), vb_ok = (tg.y != -1);
    const float vfa = va_ok ? 1.0f : 0.0f;   // pred_m = pred * vf (ref semantics)
    const float vfb = vb_ok ? 1.0f : 0.0f;

    // ---- pass 1: channel max (scalar base + shared u32 voffset, b64 loads) ----
    float ma = -3.402823466e38f, mb = -3.402823466e38f;
    #pragma unroll
    for (int c = 0; c < NC; c += 2) {
      const char* pc = sbase(pB + (size_t)c * CBYTES);   // opaque SGPR base
      float2 x0 = *(const float2*)(pc + boff);           // saddr + voffset
      float2 x1 = *(const float2*)(pc + CBYTES + boff);  // +3538944 fits imm24
      ma = fmaxf(fmaxf(ma, x0.x * vfa), x1.x * vfa);
      mb = fmaxf(fmaxf(mb, x0.y * vfb), x1.y * vfb);
    }

    // ---- prefetch next sweep (single uniform branch per iteration) ----
    if (it != ITERS - 1) {
      #pragma unroll
      for (int c = 0; c < NC; c += 2) {
        const char* pc = sbase(pB + (size_t)c * CBYTES);
        __builtin_prefetch(pc + boff + (VSWEEP / 2) * 8, 0, 1);
        __builtin_prefetch(pc + CBYTES + boff + (VSWEEP / 2) * 8, 0, 1);
      }
    }

    // ---- pass 2: sum of exp; stage e^2 (f16, 2 channels x 2 voxels) in LDS ----
    float sa = 0.f, sb = 0.f;
    #pragma unroll
    for (int p = 0; p < 44; ++p) {
      const char* pc = sbase(pB + (size_t)(2 * p) * CBYTES);
      float2 x0 = *(const float2*)(pc + boff);           // channel 2p,   voxels a,b
      float2 x1 = *(const float2*)(pc + CBYTES + boff);  // channel 2p+1, voxels a,b
      float e00 = __expf(x0.x * vfa - ma), e01 = __expf(x0.y * vfb - mb);
      float e10 = __expf(x1.x * vfa - ma), e11 = __expf(x1.y * vfb - mb);
      sa += e00 + e10; sb += e01 + e11;
      unsigned da = pk16(e00 * e00, e10 * e10);          // voxel a: (c, c+1)
      unsigned db = pk16(e01 * e01, e11 * e11);          // voxel b: (c, c+1)
      *(uint2*)&wt[p * RSTRIDE + 2 * lane] = (uint2){da, db};   // ds_store_b64
    }
    const float inva = 1.0f / sa, invb = 1.0f / sb;
    sinvd[wave][lane] = pk16(inva * inva, invb * invb);  // voxels (2*lane, 2*lane+1)

    // ---- CE + one-hot (inter/ground) for the two voxels ----
    if (va_ok) {
      const unsigned toff = (unsigned)tg.x * (unsigned)CBYTES + boff;   // < 2^31
      float xt = *(const float*)(pB + toff);
      nll_loc += (ma + __logf(sa)) - xt; vcnt_loc += 1.0f;
      atomicAdd(&s_inter[tg.x], __expf(xt - ma) * inva);
      atomicAdd(&s_ground[tg.x], 1.0f);
    }
    if (vb_ok) {
      const unsigned toff = (unsigned)tg.y * (unsigned)CBYTES + boff + 4u;
      float xt = *(const float*)(pB + toff);
      nll_loc += (mb + __logf(sb)) - xt; vcnt_loc += 1.0f;
      atomicAdd(&s_inter[tg.y], __expf(xt - mb) * invb);
      atomicAdd(&s_ground[tg.y], 1.0f);
    }

    asm volatile("s_wait_dscnt 0" ::: "memory");  // LDS in-order per wave; fence compiler

    // ---- two 32-voxel groups: build B fragments, scale by 1/s^2, accumulate ----
    #pragma unroll
    for (int g = 0; g < 2; ++g) {
      // per-lane scale vector: slot j = inv2 of voxel g*32 + hsel*16 + j
      union { uint4 q[2]; v16h h; } sc;
      {
        const unsigned int* sp = &sinvd[wave][g * 16 + hsel * 8];
        sc.q[0] = *(const uint4*)(sp + 0);
        sc.q[1] = *(const uint4*)(sp + 4);
      }
      #pragma unroll
      for (int t = 0; t < 6; ++t) {
        const int c0 = t * 16 + col;                     // this lane's channel column
        const unsigned int* src = &wt[(c0 >> 1) * RSTRIDE + g * 32 + hsel * 16];
        uint4 q0 = *(const uint4*)(src + 0);
        uint4 q1 = *(const uint4*)(src + 4);
        uint4 q2 = *(const uint4*)(src + 8);
        uint4 q3 = *(const uint4*)(src + 12);
        unsigned int dw[16] = {q0.x,q0.y,q0.z,q0.w, q1.x,q1.y,q1.z,q1.w,
                               q2.x,q2.y,q2.z,q2.w, q3.x,q3.y,q3.z,q3.w};
        union { unsigned int u[8]; v16h h; } bf;
        const bool odd = (c0 & 1);
        #pragma unroll
        for (int j = 0; j < 8; ++j) {
          unsigned int lo = dw[2*j], hi = dw[2*j+1];
          bf.u[j] = odd ? ((lo >> 16) | (hi & 0xFFFF0000u))
                        : ((lo & 0xFFFFu) | (hi << 16));
        }
        bf.h = bf.h * sc.h;                // p^2 = e^2 * inv(s)^2  (v_pk_mul_f16)
        // D[m,n] += sum_k 1 * B[k,n]  -> per-channel partial sums of p^2
        cf[t] = __builtin_amdgcn_wmma_f32_16x16x32_f16(
                    false, ones, false, bf.h, (short)0, cf[t], false, false);
      }
    }
  }

  // ---- drain per-wave WMMA accumulators into block accumulators ----
  #pragma unroll
  for (int t = 0; t < 6; ++t) {
    const int ch = t * 16 + col;            // lanes 0..15 carry unique columns
    if (lane < 16 && ch < NC) atomicAdd(&s_predo[ch], cf[t][0]);
  }
  atomicAdd(&s_nll, nll_loc);
  atomicAdd(&s_vcnt, vcnt_loc);
  __syncthreads();

  // ---- one global atomic per block per channel ----
  if (tid < NC) {
    atomicAdd(&ws[WS_PREDO  + b * NC + tid], s_predo[tid]);
    atomicAdd(&ws[WS_INTER  + b * NC + tid], s_inter[tid]);
    atomicAdd(&ws[WS_GROUND + b * NC + tid], s_ground[tid]);
  }
  if (tid == 0) {
    atomicAdd(&ws[WS_NLL],  s_nll);
    atomicAdd(&ws[WS_VCNT], s_vcnt);
  }
}

__global__ __launch_bounds__(256)
void loss_final(const float* __restrict__ ws, float* __restrict__ out) {
  __shared__ float red[256];
  const int tid = threadIdx.x;
  float d = 0.f;
  if (tid < NB * NC) {
    float po = ws[WS_PREDO + tid], in = ws[WS_INTER + tid], go = ws[WS_GROUND + tid];
    d = 1.0f - (2.0f * in + SMOOTH) / (go + po + SMOOTH);
  }
  red[tid] = d;
  __syncthreads();
  #pragma unroll
  for (int s = 128; s > 0; s >>= 1) {
    if (tid < s) red[tid] += red[tid + s];
    __syncthreads();
  }
  if (tid == 0) {
    float dice = red[0] / (float)(NB * NC);
    float ce   = ws[WS_NLL] / fmaxf(ws[WS_VCNT], 1.0f);
    out[0] = CE_W * ce + DICE_W * dice;
  }
}

extern "C" void kernel_launch(void* const* d_in, const int* in_sizes, int n_in,
                              void* d_out, int out_size, void* d_ws, size_t ws_size,
                              hipStream_t stream) {
  const float* pred   = (const float*)d_in[0];
  const int*   target = (const int*)d_in[1];
  float*       out    = (float*)d_out;
  float*       ws     = (float*)d_ws;

  (void)hipMemsetAsync(ws, 0, 530 * sizeof(float), stream);
  dim3 grid(GX, NB, 1);
  loss_main<<<grid, BLK, 0, stream>>>(pred, target, ws);
  loss_final<<<1, 256, 0, stream>>>(ws, out);
}
